// EncodedGINE_38233798869093
// MI455X (gfx1250) — compile-verified, matrix operand order checked by reading, not run on hardware
//
#include <hip/hip_runtime.h>
#include <hip/hip_bf16.h>
#include <cstdint>

// ---------------------------------------------------------------------------
// Problem constants (from the reference)
// ---------------------------------------------------------------------------
#define NN     32768     // nodes
#define EE     65536     // edges
#define DD     256       // hidden dim
#define HH     4         // heads
#define GG     2048      // graphs
#define TT     12        // targets
#define LL     4         // conv layers
#define SN     9         // node features / tokens
#define SE     3         // edge features / tokens
#define NVN    119       // node vocab
#define NVE    22        // edge vocab

typedef __attribute__((ext_vector_type(16))) __bf16        v16bf;
typedef __attribute__((ext_vector_type(8)))  float         v8f;
typedef __attribute__((ext_vector_type(4)))  unsigned int  u32x4;
typedef __attribute__((ext_vector_type(4)))  int           i32x4;
typedef __attribute__((ext_vector_type(8)))  int           i32x8;

#if defined(__has_builtin)
#if __has_builtin(__builtin_amdgcn_tensor_load_to_lds)
#define HAVE_TDM 1
#endif
#endif

union FragBF {            // one WMMA 16-bf16 operand fragment (8 VGPRs)
    v16bf v;
    u32x4 q[2];
};

// ---------------------------------------------------------------------------
// Small utility kernels
// ---------------------------------------------------------------------------
__global__ void zero_f32(float* __restrict__ p, int n) {
    int i = blockIdx.x * blockDim.x + threadIdx.x;
    if (i < n) p[i] = 0.f;
}

__global__ void cvt_f32_bf16(const float* __restrict__ in, __bf16* __restrict__ out, int n) {
    int i = blockIdx.x * blockDim.x + threadIdx.x;
    if (i < n) out[i] = (__bf16)in[i];
}

// z = (1 + eps[l]) * h
__global__ void scale_init(const float* __restrict__ h, const float* __restrict__ eps,
                           int l, float* __restrict__ z, int n) {
    int i = blockIdx.x * blockDim.x + threadIdx.x;
    if (i < n) z[i] = (1.f + eps[l]) * h[i];
}

// msg = relu(h[src] + e_l);  z[dst] += msg   (4 floats / thread)
__global__ void scatter_edges(const float* __restrict__ el, const float* __restrict__ h,
                              const int* __restrict__ ei, float* __restrict__ z) {
    int idx = blockIdx.x * blockDim.x + threadIdx.x;
    if (idx >= EE * 64) return;
    int e = idx >> 6;
    int d = (idx & 63) * 4;
    int src = ei[e];
    int dst = ei[EE + e];
    const float4 ev = *(const float4*)(el + (size_t)e * DD + d);
    const float4 hv = *(const float4*)(h + (size_t)src * DD + d);
    float m0 = fmaxf(ev.x + hv.x, 0.f);
    float m1 = fmaxf(ev.y + hv.y, 0.f);
    float m2 = fmaxf(ev.z + hv.z, 0.f);
    float m3 = fmaxf(ev.w + hv.w, 0.f);
    float* zp = z + (size_t)dst * DD + d;
    unsafeAtomicAdd(zp + 0, m0);
    unsafeAtomicAdd(zp + 1, m1);
    unsafeAtomicAdd(zp + 2, m2);
    unsafeAtomicAdd(zp + 3, m3);
}

// pooled[batch[n]] += h[n]   (4 floats / thread)
__global__ void scatter_pool(const float* __restrict__ h, const int* __restrict__ batch,
                             float* __restrict__ pooled) {
    int idx = blockIdx.x * blockDim.x + threadIdx.x;
    if (idx >= NN * 64) return;
    int node = idx >> 6;
    int d = (idx & 63) * 4;
    int g = batch[node];
    const float4 hv = *(const float4*)(h + (size_t)node * DD + d);
    float* pp = pooled + (size_t)g * DD + d;
    unsafeAtomicAdd(pp + 0, hv.x);
    unsafeAtomicAdd(pp + 1, hv.y);
    unsafeAtomicAdd(pp + 2, hv.z);
    unsafeAtomicAdd(pp + 3, hv.w);
}

// per-channel sum / sumsq over M rows (row-major M x 256)
__global__ void bn_stats(const float* __restrict__ x, float* __restrict__ sums, int rowsPerBlock) {
    int c = threadIdx.x;                            // 256 channels
    size_t r0 = (size_t)blockIdx.x * rowsPerBlock;
    float s = 0.f, ss = 0.f;
    for (int i = 0; i < rowsPerBlock; ++i) {
        float v = x[(r0 + i) * DD + c];
        s += v;
        ss += v * v;
    }
    unsafeAtomicAdd(&sums[c], s);
    unsafeAtomicAdd(&sums[DD + c], ss);
}

// out = relu(batchnorm(x)) (+ residual)
__global__ void bn_apply(const float* __restrict__ x, const float* __restrict__ sums,
                         const float* __restrict__ g, const float* __restrict__ b,
                         const float* __restrict__ residual, float* __restrict__ outf, int M) {
    int i = blockIdx.x * blockDim.x + threadIdx.x;
    if (i >= M * DD) return;
    int c = i & (DD - 1);
    float mean = sums[c] / (float)M;
    float var  = sums[DD + c] / (float)M - mean * mean;
    float v = (x[i] - mean) * rsqrtf(var + 1e-5f) * g[c] + b[c];
    v = fmaxf(v, 0.f);
    if (residual) v += residual[i];
    outf[i] = v;
}

// out = p2 @ W2^T + b2   (G x 12, tiny)
__global__ void final_proj(const float* __restrict__ p2, const float* __restrict__ W2,
                           const float* __restrict__ b2, float* __restrict__ out) {
    int i = blockIdx.x * blockDim.x + threadIdx.x;
    if (i >= GG * TT) return;
    int g = i / TT, t = i - g * TT;
    const float* pr = p2 + (size_t)g * DD;
    const float* wr = W2 + (size_t)t * DD;
    float s = b2[t];
    #pragma unroll 4
    for (int k = 0; k < DD; ++k) s += pr[k] * wr[k];
    out[i] = s;
}

// ---------------------------------------------------------------------------
// WMMA GEMM:  C[M x 256] = A[M x 256] (bf16) * W[256 x 256]^T (bf16) + bias
// grid = (M/128, 4), block = 128 (4 waves).
//   - W tile (64 rows x 256 = 32 KB, contiguous) staged in LDS via the
//     Tensor Data Mover (TENSOR_LOAD_TO_LDS, waited with s_wait_tensorcnt),
//     cooperative-copy fallback if the builtin is unavailable.
//   - wave w owns rows [bx*128 + 32w, +32) (2 M-subtiles) x 64 cols
//     (4 N-subtiles) -> 8 independent WMMA chains per k-step; B fragments
//     from LDS, A fragments from global (b128), A reused across 4 subtiles.
// ---------------------------------------------------------------------------
__global__ __launch_bounds__(128) void gemm256_wmma(
    const __bf16* __restrict__ A, const __bf16* __restrict__ W,
    const float* __restrict__ bias, float* __restrict__ outf,
    __bf16* __restrict__ outb, int relu)
{
    __shared__ __align__(16) __bf16 s_B[64][DD];    // 32 KB weight tile

    const int tid  = threadIdx.x;
    const int wave = tid >> 5;
    const int lane = tid & 31;
    const int hl   = lane >> 4;     // which 16-lane half
    const int lr   = lane & 15;     // lane-in-half == row/col in tile
    const int rowM = blockIdx.x * 128 + wave * 32;
    const int colN = blockIdx.y * 64;

    // ---- stage the 32 KB contiguous weight tile into LDS ----
#ifdef HAVE_TDM
    if (wave == 0) {
        // D# group 0: count=1, lds_addr, global_addr, type=2 ("image")
        uint32_t lds_base = (uint32_t)(uintptr_t)&s_B[0][0];
        uint64_t ga = (uint64_t)(uintptr_t)(W + (size_t)colN * DD);
        u32x4 g0;
        g0[0] = 1u;                                             // count=1
        g0[1] = lds_base;                                       // lds_addr
        g0[2] = (uint32_t)ga;                                   // global_addr lo
        g0[3] = (uint32_t)((ga >> 32) & 0x1FFFFFFu) | (2u << 30); // ga hi | type=2
        // D# group 1: data_size=8B; 1-D tile: 4096 x 8B = 32 KB
        i32x8 g1;
        g1[0] = 0x00030000;          // data_size=3 (8 bytes), no multicast/pad
        g1[1] = (int)(0x1000u << 16);// tensor_dim0[15:0]=4096 at bits[63:48]
        g1[2] = (int)(1u << 16);     // tensor_dim0 hi=0 | tensor_dim1=1
        g1[3] = (int)(0x1000u << 16);// tensor_dim1 hi=0 | tile_dim0=4096
        g1[4] = 1;                   // tile_dim1=1, tile_dim2=0
        g1[5] = 0x1000;              // tensor_dim0_stride=4096
        g1[6] = 0;
        g1[7] = 0;
        i32x4 gz;
        gz[0] = 0; gz[1] = 0; gz[2] = 0; gz[3] = 0;
        i32x8 gz8;
        gz8[0] = 0; gz8[1] = 0; gz8[2] = 0; gz8[3] = 0;
        gz8[4] = 0; gz8[5] = 0; gz8[6] = 0; gz8[7] = 0;
        __builtin_amdgcn_tensor_load_to_lds(g0, g1, gz, gz, gz8, 0);
        __builtin_amdgcn_s_wait_tensorcnt(0);
    }
#else
    {
        const u32x4* src = (const u32x4*)(W + (size_t)colN * DD);
        u32x4* dst = (u32x4*)&s_B[0][0];
        for (int i = tid; i < 64 * DD / 8; i += 128) dst[i] = src[i];
    }
#endif
    __syncthreads();

    const __bf16* arow0 = A + (size_t)(rowM + lr) * DD;
    const __bf16* arow1 = A + (size_t)(rowM + 16 + lr) * DD;

    v8f acc[2][4];
    #pragma unroll
    for (int mt = 0; mt < 2; ++mt)
        #pragma unroll
        for (int s = 0; s < 4; ++s)
            #pragma unroll
            for (int r = 0; r < 8; ++r) acc[mt][s][r] = 0.f;

    #pragma unroll
    for (int kb = 0; kb < DD; kb += 32) {
        FragBF a0, a1;
        a0.q[0] = *(const u32x4*)(arow0 + kb + 8 * hl);
        a0.q[1] = *(const u32x4*)(arow0 + kb + 16 + 8 * hl);
        a1.q[0] = *(const u32x4*)(arow1 + kb + 8 * hl);
        a1.q[1] = *(const u32x4*)(arow1 + kb + 16 + 8 * hl);
        if (kb + 64 < DD) {
            __builtin_prefetch(arow0 + kb + 64, 0, 1);
            __builtin_prefetch(arow1 + kb + 64, 0, 1);
        }
        #pragma unroll
        for (int sub = 0; sub < 4; ++sub) {
            FragBF b;
            b.q[0] = *(const u32x4*)&s_B[sub * 16 + lr][kb + 8 * hl];
            b.q[1] = *(const u32x4*)&s_B[sub * 16 + lr][kb + 16 + 8 * hl];
            acc[0][sub] = __builtin_amdgcn_wmma_f32_16x16x32_bf16(
                false, a0.v, false, b.v, (short)0, acc[0][sub], false, false);
            acc[1][sub] = __builtin_amdgcn_wmma_f32_16x16x32_bf16(
                false, a1.v, false, b.v, (short)0, acc[1][sub], false, false);
        }
    }

    #pragma unroll
    for (int mt = 0; mt < 2; ++mt) {
        #pragma unroll
        for (int sub = 0; sub < 4; ++sub) {
            int n = colN + sub * 16 + lr;
            float bn = bias ? bias[n] : 0.f;
            #pragma unroll
            for (int r = 0; r < 8; ++r) {
                int m = rowM + mt * 16 + 8 * hl + r;
                float v = acc[mt][sub][r] + bn;
                if (relu) v = fmaxf(v, 0.f);
                if (outf) outf[(size_t)m * DD + n] = v;
                if (outb) outb[(size_t)m * DD + n] = (__bf16)v;
            }
        }
    }
}

// ---------------------------------------------------------------------------
// Fused feature encoder (1 block = 1 node/edge, 128 threads = 4 waves).
//   seq  = emb[s, ids[node,s]]                 (S x 256)
//   qkv  = seq @ Wqkv^T + bqkv                 (WMMA, S rows padded to 16)
//   attn = per-head softmax(q k^T / 8) v       (VALU, wave w = head w)
//   post = attn @ Wo^T + bo + seq              (WMMA)
//   out  = mean_s layernorm(post)
// ---------------------------------------------------------------------------
template <int S, int NV>
__global__ __launch_bounds__(128) void encoder_kernel(
    const int* __restrict__ ids, const float* __restrict__ emb,
    const __bf16* __restrict__ Wqkv, const float* __restrict__ bqkv,
    const __bf16* __restrict__ Wo, const float* __restrict__ bo,
    const float* __restrict__ ln_g, const float* __restrict__ ln_b,
    float* __restrict__ hf, __bf16* __restrict__ hb)
{
    __shared__ int s_ids[16];
    __shared__ __align__(16) __bf16 s_seq[16][DD];         // bf16 tokens (padded)
    __shared__ __align__(16) float  s_seqf[S][DD];         // f32 tokens (residual)
    __shared__ __align__(16) __bf16 s_qkv[16][3 * DD];     // q|k|v per row
    __shared__ __align__(16) __bf16 s_att[16][DD];         // attention output
    __shared__ __align__(16) float  s_post[S][DD];         // pre-LN
    __shared__ float s_mean[S], s_rstd[S];

    const int node = blockIdx.x;
    const int tid  = threadIdx.x;

    if (tid < S) {
        int id = ids[(size_t)node * S + tid];
        id = id < 0 ? 0 : (id > NV - 1 ? NV - 1 : id);
        s_ids[tid] = id;
    }
    __syncthreads();

    // gather embedding rows; zero-pad rows >= S
    for (int e = tid; e < 16 * DD; e += 128) {
        int row = e >> 8, d = e & (DD - 1);
        float v = 0.f;
        if (row < S) {
            v = emb[((size_t)row * NV + s_ids[row]) * DD + d];
            s_seqf[row][d] = v;
        }
        s_seq[row][d] = (__bf16)v;
        s_att[row][d] = (__bf16)0.f;
    }
    __syncthreads();

    const int wave = tid >> 5;
    const int lane = tid & 31;
    const int hl   = lane >> 4;
    const int lr   = lane & 15;

    // ---- QKV projection: 48 column tiles of 16 over 3*D = 768 ----
    for (int ct = wave; ct < 48; ct += 4) {
        v8f acc;
        #pragma unroll
        for (int r = 0; r < 8; ++r) acc[r] = 0.f;
        const __bf16* wrow = Wqkv + (size_t)(ct * 16 + lr) * DD;
        #pragma unroll
        for (int kb = 0; kb < DD; kb += 32) {
            FragBF a, b;
            a.q[0] = *(const u32x4*)&s_seq[lr][kb + 8 * hl];
            a.q[1] = *(const u32x4*)&s_seq[lr][kb + 16 + 8 * hl];
            b.q[0] = *(const u32x4*)(wrow + kb + 8 * hl);
            b.q[1] = *(const u32x4*)(wrow + kb + 16 + 8 * hl);
            acc = __builtin_amdgcn_wmma_f32_16x16x32_bf16(
                false, a.v, false, b.v, (short)0, acc, false, false);
        }
        int n = ct * 16 + lr;
        float bn = bqkv[n];
        #pragma unroll
        for (int r = 0; r < 8; ++r) {
            int m = 8 * hl + r;
            s_qkv[m][n] = (__bf16)(acc[r] + bn);
        }
    }
    __syncthreads();

    // ---- attention: wave = head; lanes act as t for scores, as d for output
    {
        const int head = wave;              // HH == 4 waves
        const int base = head * 64;
        for (int s = 0; s < S; ++s) {
            float sc = -1e30f;
            if (lane < S) {
                float sum = 0.f;
                #pragma unroll
                for (int d = 0; d < 64; ++d)
                    sum += (float)s_qkv[s][base + d] * (float)s_qkv[lane][DD + base + d];
                sc = sum * 0.125f;          // 1/sqrt(64)
            }
            float mx = sc;
            for (int m = 16; m > 0; m >>= 1) mx = fmaxf(mx, __shfl_xor(mx, m));
            float ex = (lane < S) ? __expf(sc - mx) : 0.f;
            float den = ex;
            for (int m = 16; m > 0; m >>= 1) den += __shfl_xor(den, m);
            float wt_l = ex / den;
            float o0 = 0.f, o1 = 0.f;
            for (int t = 0; t < S; ++t) {
                float wt = __shfl(wt_l, t);
                o0 += wt * (float)s_qkv[t][2 * DD + base + lane];
                o1 += wt * (float)s_qkv[t][2 * DD + base + 32 + lane];
            }
            s_att[s][base + lane]      = (__bf16)o0;
            s_att[s][base + 32 + lane] = (__bf16)o1;
        }
    }
    __syncthreads();

    // ---- output projection + residual: 16 column tiles over D ----
    for (int ct = wave; ct < 16; ct += 4) {
        v8f acc;
        #pragma unroll
        for (int r = 0; r < 8; ++r) acc[r] = 0.f;
        const __bf16* wrow = Wo + (size_t)(ct * 16 + lr) * DD;
        #pragma unroll
        for (int kb = 0; kb < DD; kb += 32) {
            FragBF a, b;
            a.q[0] = *(const u32x4*)&s_att[lr][kb + 8 * hl];
            a.q[1] = *(const u32x4*)&s_att[lr][kb + 16 + 8 * hl];
            b.q[0] = *(const u32x4*)(wrow + kb + 8 * hl);
            b.q[1] = *(const u32x4*)(wrow + kb + 16 + 8 * hl);
            acc = __builtin_amdgcn_wmma_f32_16x16x32_bf16(
                false, a.v, false, b.v, (short)0, acc, false, false);
        }
        int n = ct * 16 + lr;
        float bn = bo[n];
        #pragma unroll
        for (int r = 0; r < 8; ++r) {
            int m = 8 * hl + r;
            if (m < S) s_post[m][n] = acc[r] + bn + s_seqf[m][n];
        }
    }
    __syncthreads();

    // ---- layernorm stats per token row ----
    for (int s = wave; s < S; s += 4) {
        float sum = 0.f, ss = 0.f;
        #pragma unroll
        for (int i = 0; i < 8; ++i) {
            float v = s_post[s][lane + 32 * i];
            sum += v;
            ss  += v * v;
        }
        for (int m = 16; m > 0; m >>= 1) {
            sum += __shfl_xor(sum, m);
            ss  += __shfl_xor(ss, m);
        }
        if (lane == 0) {
            float mean = sum / (float)DD;
            float var  = ss / (float)DD - mean * mean;
            s_mean[s] = mean;
            s_rstd[s] = rsqrtf(var + 1e-5f);
        }
    }
    __syncthreads();

    // ---- mean over S of layernorm rows ----
    for (int d = tid; d < DD; d += 128) {
        float acc = 0.f;
        #pragma unroll
        for (int s = 0; s < S; ++s)
            acc += (s_post[s][d] - s_mean[s]) * s_rstd[s];
        float v = ln_g[d] * (acc / (float)S) + ln_b[d];
        if (hf) hf[(size_t)node * DD + d] = v;
        if (hb) hb[(size_t)node * DD + d] = (__bf16)v;
    }
}

// ---------------------------------------------------------------------------
// Orchestration
// ---------------------------------------------------------------------------
extern "C" void kernel_launch(void* const* d_in, const int* in_sizes, int n_in,
                              void* d_out, int out_size, void* d_ws, size_t ws_size,
                              hipStream_t stream) {
    (void)in_sizes; (void)n_in; (void)out_size; (void)ws_size;

    const int*   x          = (const int*)d_in[0];
    const int*   edge_attr  = (const int*)d_in[1];
    const int*   edge_index = (const int*)d_in[2];
    const int*   batch      = (const int*)d_in[3];
    const float* node_emb   = (const float*)d_in[4];
    const float* node_Wqkv  = (const float*)d_in[5];
    const float* node_bqkv  = (const float*)d_in[6];
    const float* node_Wo    = (const float*)d_in[7];
    const float* node_bo    = (const float*)d_in[8];
    const float* node_ln_g  = (const float*)d_in[9];
    const float* node_ln_b  = (const float*)d_in[10];
    const float* edge_emb   = (const float*)d_in[11];
    const float* edge_Wqkv  = (const float*)d_in[12];
    const float* edge_bqkv  = (const float*)d_in[13];
    const float* edge_Wo    = (const float*)d_in[14];
    const float* edge_bo    = (const float*)d_in[15];
    const float* edge_ln_g  = (const float*)d_in[16];
    const float* edge_ln_b  = (const float*)d_in[17];
    const float* conv_linW  = (const float*)d_in[18];
    const float* conv_linb  = (const float*)d_in[19];
    const float* conv_W1    = (const float*)d_in[20];
    const float* conv_b1    = (const float*)d_in[21];
    const float* conv_W2    = (const float*)d_in[22];
    const float* conv_b2    = (const float*)d_in[23];
    const float* conv_eps   = (const float*)d_in[24];
    const float* bn_g       = (const float*)d_in[25];
    const float* bn_b       = (const float*)d_in[26];
    const float* out_W1     = (const float*)d_in[27];
    const float* out_b1     = (const float*)d_in[28];
    const float* out_bn_g   = (const float*)d_in[29];
    const float* out_bn_b   = (const float*)d_in[30];
    const float* out_W2     = (const float*)d_in[31];
    const float* out_b2     = (const float*)d_in[32];

    // workspace carve-up (deterministic every call)
    size_t off = 0;
    auto alloc = [&](size_t bytes) -> void* {
        void* p = (char*)d_ws + off;
        off += (bytes + 255) & ~(size_t)255;
        return p;
    };
    __bf16* nWqkv_bf = (__bf16*)alloc((size_t)3 * DD * DD * 2);
    __bf16* nWo_bf   = (__bf16*)alloc((size_t)DD * DD * 2);
    __bf16* eWqkv_bf = (__bf16*)alloc((size_t)3 * DD * DD * 2);
    __bf16* eWo_bf   = (__bf16*)alloc((size_t)DD * DD * 2);
    __bf16* linW_bf  = (__bf16*)alloc((size_t)LL * DD * DD * 2);
    __bf16* W1_bf    = (__bf16*)alloc((size_t)LL * DD * DD * 2);
    __bf16* W2_bf    = (__bf16*)alloc((size_t)LL * DD * DD * 2);
    __bf16* oW1_bf   = (__bf16*)alloc((size_t)DD * DD * 2);
    float*  h        = (float*)alloc((size_t)NN * DD * 4);
    __bf16* e_bf     = (__bf16*)alloc((size_t)EE * DD * 2);
    float*  e_l      = (float*)alloc((size_t)EE * DD * 4);
    float*  z        = (float*)alloc((size_t)NN * DD * 4);
    __bf16* z_bf     = (__bf16*)alloc((size_t)NN * DD * 2);
    __bf16* z1_bf    = (__bf16*)alloc((size_t)NN * DD * 2);
    float*  z2       = (float*)alloc((size_t)NN * DD * 4);
    float*  bnsum    = (float*)alloc((size_t)2 * DD * 4);
    float*  pooled   = (float*)alloc((size_t)GG * DD * 4);
    __bf16* pooled_bf= (__bf16*)alloc((size_t)GG * DD * 2);
    float*  p1       = (float*)alloc((size_t)GG * DD * 4);
    float*  p2       = (float*)alloc((size_t)GG * DD * 4);

    auto cvt = [&](const float* src, __bf16* dst, int n) {
        cvt_f32_bf16<<<(n + 255) / 256, 256, 0, stream>>>(src, dst, n);
    };

    // ---- weight conversion to bf16 ----
    cvt(node_Wqkv, nWqkv_bf, 3 * DD * DD);
    cvt(node_Wo,   nWo_bf,   DD * DD);
    cvt(edge_Wqkv, eWqkv_bf, 3 * DD * DD);
    cvt(edge_Wo,   eWo_bf,   DD * DD);
    cvt(conv_linW, linW_bf,  LL * DD * DD);
    cvt(conv_W1,   W1_bf,    LL * DD * DD);
    cvt(conv_W2,   W2_bf,    LL * DD * DD);
    cvt(out_W1,    oW1_bf,   DD * DD);

    // ---- feature encoders ----
    encoder_kernel<SN, NVN><<<NN, 128, 0, stream>>>(
        x, node_emb, nWqkv_bf, node_bqkv, nWo_bf, node_bo,
        node_ln_g, node_ln_b, h, (__bf16*)nullptr);
    encoder_kernel<SE, NVE><<<EE, 128, 0, stream>>>(
        edge_attr, edge_emb, eWqkv_bf, edge_bqkv, eWo_bf, edge_bo,
        edge_ln_g, edge_ln_b, (float*)nullptr, e_bf);

    // ---- GINE conv layers ----
    for (int l = 0; l < LL; ++l) {
        gemm256_wmma<<<dim3(EE / 128, 4), 128, 0, stream>>>(
            e_bf, linW_bf + (size_t)l * DD * DD, conv_linb + l * DD,
            e_l, (__bf16*)nullptr, 0);
        scale_init<<<(NN * DD) / 256, 256, 0, stream>>>(h, conv_eps, l, z, NN * DD);
        scatter_edges<<<(EE * 64) / 256, 256, 0, stream>>>(e_l, h, edge_index, z);
        cvt(z, z_bf, NN * DD);
        gemm256_wmma<<<dim3(NN / 128, 4), 128, 0, stream>>>(
            z_bf, W1_bf + (size_t)l * DD * DD, conv_b1 + l * DD,
            (float*)nullptr, z1_bf, 1);
        gemm256_wmma<<<dim3(NN / 128, 4), 128, 0, stream>>>(
            z1_bf, W2_bf + (size_t)l * DD * DD, conv_b2 + l * DD,
            z2, (__bf16*)nullptr, 1);
        zero_f32<<<2, 256, 0, stream>>>(bnsum, 2 * DD);
        bn_stats<<<NN / 128, 256, 0, stream>>>(z2, bnsum, 128);
        bn_apply<<<(NN * DD) / 256, 256, 0, stream>>>(
            z2, bnsum, bn_g + l * DD, bn_b + l * DD, h, h, NN);
    }

    // ---- readout ----
    zero_f32<<<(GG * DD) / 256, 256, 0, stream>>>(pooled, GG * DD);
    scatter_pool<<<(NN * 64) / 256, 256, 0, stream>>>(h, batch, pooled);
    cvt(pooled, pooled_bf, GG * DD);
    gemm256_wmma<<<dim3(GG / 128, 4), 128, 0, stream>>>(
        pooled_bf, oW1_bf, out_b1, p1, (__bf16*)nullptr, 0);
    zero_f32<<<2, 256, 0, stream>>>(bnsum, 2 * DD);
    bn_stats<<<GG / 128, 256, 0, stream>>>(p1, bnsum, 128);
    bn_apply<<<(GG * DD) / 256, 256, 0, stream>>>(
        p1, bnsum, out_bn_g, out_bn_b, (float*)nullptr, p2, GG);
    final_proj<<<(GG * TT + 255) / 256, 256, 0, stream>>>(p2, out_W2, out_b2, (float*)d_out);
}
